// MultiHeadAttention_87660282511883
// MI455X (gfx1250) — matrix operand order repaired and checked by reading, not compiled, not run on hardware
//
#include <hip/hip_runtime.h>
#include <hip/hip_bf16.h>
#include <cstdint>
#include <cstddef>

// MI455X / gfx1250 MHA forward.
// All GEMMs use v_wmma_f32_16x16x32_bf16 (wave32, 16x16 tiles, f32 accum).
// Conversion-free tile stages use global_load_async_to_lds (ASYNCcnt) when
// the toolchain exposes the builtins.

#define BDIM 256

typedef __attribute__((ext_vector_type(16))) __bf16          v16bf;
typedef __attribute__((ext_vector_type(16))) unsigned short  v16u;
typedef __attribute__((ext_vector_type(8)))  float           v8f;
typedef __attribute__((ext_vector_type(4)))  int             v4i;

static constexpr int BATCH  = 2;
static constexpr int SEQ    = 2048;
static constexpr int DMODEL = 1024;
static constexpr int NHEADS = 16;
static constexpr int DHEAD  = 64;

#if defined(__has_builtin)
#if __has_builtin(__builtin_amdgcn_global_load_async_to_lds_b128) && \
    __has_builtin(__builtin_amdgcn_s_wait_asynccnt)
#define USE_ASYNC_LDS 1
#endif
#endif

#ifdef USE_ASYNC_LDS
__device__ __forceinline__ void async_cp16(void* lds, const void* g) {
  __builtin_amdgcn_global_load_async_to_lds_b128(
      (__attribute__((address_space(1))) v4i*)g,
      (__attribute__((address_space(3))) v4i*)lds, 0, 0);
}
__device__ __forceinline__ void async_fence() {
  __builtin_amdgcn_s_wait_asynccnt(0);
}
#else
__device__ __forceinline__ void async_cp16(void* lds, const void* g) {
  *(uint4*)lds = *(const uint4*)g;
}
__device__ __forceinline__ void async_fence() {}
#endif

__device__ __forceinline__ unsigned short f32_to_bf16(float f) {
  unsigned int u = __builtin_bit_cast(unsigned int, f);
  u += 0x7FFFu + ((u >> 16) & 1u);          // round-to-nearest-even
  return (unsigned short)(u >> 16);
}

__device__ __forceinline__ v16bf ld_frag(const unsigned short* p0, const unsigned short* p1) {
  union { v16u v; uint4 q[2]; } u;
  u.q[0] = *(const uint4*)p0;               // ds_load_b128
  u.q[1] = *(const uint4*)p1;
  return __builtin_bit_cast(v16bf, u.v);
}

__device__ __forceinline__ v16bf ld_fragT(const unsigned short* p, int ld) {
  v16u v;
#pragma unroll
  for (int s = 0; s < 8; ++s) { v[s] = p[s * ld]; v[s + 8] = p[(16 + s) * ld]; }
  return __builtin_bit_cast(v16bf, v);
}

__device__ __forceinline__ v8f wmma_bf16(v16bf a, v16bf b, v8f c) {
  return __builtin_amdgcn_wmma_f32_16x16x32_bf16(false, a, false, b, (short)0, c, false, false);
}

__device__ __forceinline__ v8f zero8() {
  v8f z = {0.f, 0.f, 0.f, 0.f, 0.f, 0.f, 0.f, 0.f};
  return z;
}

// ---------------------------------------------------------------------------
// C[4096][1024] = A[4096][1024] * W[1024][1024]^T   (both operands K-major)
// EPI 0: write bf16 into head-major [B,H,S,64].  EPI 1: write fp32 pre-LN.
// ---------------------------------------------------------------------------
template<bool A_BF16, int EPI>
__global__ __launch_bounds__(BDIM)
void gemm_nt_kernel(const void* __restrict__ Av, const float* __restrict__ W,
                    void* __restrict__ dst) {
  constexpr int K   = 1024;
  constexpr int LDA = 48;    // ushorts per row (96 B, 16-B aligned)
  constexpr int LDB = 136;   // transposed [k][n] (272 B rows)
  __shared__ __align__(16) unsigned short lsA[128 * LDA];
  __shared__ __align__(16) unsigned short lsBT[32 * LDB];

  const int tid   = threadIdx.x;
  const int lane  = tid & 31;         // wave32
  const int wave  = tid >> 5;         // 8 waves
  const int half  = lane >> 4;
  const int mlane = lane & 15;
  const int wm    = (wave & 3) * 32;  // 4 waves in M
  const int wn    = (wave >> 2) * 64; // 2 waves in N
  const int mBlk  = blockIdx.y * 128;
  const int nBlk  = blockIdx.x * 128;

  v8f acc[2][4];
#pragma unroll
  for (int t = 0; t < 2; ++t)
#pragma unroll
    for (int j = 0; j < 4; ++j) acc[t][j] = zero8();

  for (int kt = 0; kt < K / 32; ++kt) {
    const int k0 = kt * 32;
    // ---- stage A tile (128 x 32) into LDS as bf16, row-major [m][k] ----
    if constexpr (A_BF16) {
      const unsigned short* A = (const unsigned short*)Av;
#pragma unroll
      for (int i = 0; i < 2; ++i) {
        int lin = tid + i * BDIM;
        int row = lin >> 2, g = lin & 3;
        async_cp16(&lsA[row * LDA + g * 8],
                   A + (size_t)(mBlk + row) * K + k0 + g * 8);
      }
    } else {
      const float* A = (const float*)Av;
#pragma unroll
      for (int i = 0; i < 4; ++i) {
        int lin = tid + i * BDIM;
        int row = lin >> 3, cg = lin & 7;
        float4 v = *(const float4*)(A + (size_t)(mBlk + row) * K + k0 + cg * 4);
        uint2 p;
        p.x = (unsigned)f32_to_bf16(v.x) | ((unsigned)f32_to_bf16(v.y) << 16);
        p.y = (unsigned)f32_to_bf16(v.z) | ((unsigned)f32_to_bf16(v.w) << 16);
        *(uint2*)&lsA[row * LDA + cg * 4] = p;
      }
    }
    // ---- stage W tile transposed into LDS: lsBT[k][n] ----
#pragma unroll
    for (int i = 0; i < 4; ++i) {
      int lin = tid + i * BDIM;
      int n = lin >> 3, cg = lin & 7;
      float4 v = *(const float4*)(W + (size_t)(nBlk + n) * K + k0 + cg * 4);
      lsBT[(cg * 4 + 0) * LDB + n] = f32_to_bf16(v.x);
      lsBT[(cg * 4 + 1) * LDB + n] = f32_to_bf16(v.y);
      lsBT[(cg * 4 + 2) * LDB + n] = f32_to_bf16(v.z);
      lsBT[(cg * 4 + 3) * LDB + n] = f32_to_bf16(v.w);
    }
    if (kt + 1 < K / 32) {  // gfx1250 global_prefetch_b8 for next K tile of W
      __builtin_prefetch(W + (size_t)(nBlk + (tid >> 1)) * K + k0 + 32, 0, 3);
    }
    if constexpr (A_BF16) async_fence();
    __syncthreads();

    // ---- WMMA: 2 M-tiles x 4 N-tiles per wave ----
    const unsigned short* a0p = lsA + (wm + mlane) * LDA + half * 8;
    const unsigned short* a1p = lsA + (wm + 16 + mlane) * LDA + half * 8;
    v16bf a0 = ld_frag(a0p, a0p + 16);
    v16bf a1 = ld_frag(a1p, a1p + 16);
    const unsigned short* bp = lsBT + half * 8 * LDB + wn + mlane;   // lane <-> N col, elems over K
    v16bf b0 = ld_fragT(bp,      LDB);
    v16bf b1 = ld_fragT(bp + 16, LDB);
    v16bf b2 = ld_fragT(bp + 32, LDB);
    v16bf b3 = ld_fragT(bp + 48, LDB);
    acc[0][0] = wmma_bf16(a0, b0, acc[0][0]);
    acc[0][1] = wmma_bf16(a0, b1, acc[0][1]);
    acc[0][2] = wmma_bf16(a0, b2, acc[0][2]);
    acc[0][3] = wmma_bf16(a0, b3, acc[0][3]);
    acc[1][0] = wmma_bf16(a1, b0, acc[1][0]);
    acc[1][1] = wmma_bf16(a1, b1, acc[1][1]);
    acc[1][2] = wmma_bf16(a1, b2, acc[1][2]);
    acc[1][3] = wmma_bf16(a1, b3, acc[1][3]);
    __syncthreads();
  }

  // ---- epilogue: C/D layout lane->N, VGPR r -> M = r + 8*half ----
#pragma unroll
  for (int t = 0; t < 2; ++t)
#pragma unroll
    for (int r = 0; r < 8; ++r) {
      const int gm = mBlk + wm + t * 16 + r + 8 * half;
      if constexpr (EPI == 0) {  // bf16, head-major [B,H,S,64]
        const int bb = gm >> 11, s = gm & (SEQ - 1);
        unsigned short* rowDst = (unsigned short*)dst +
            ((size_t)bb * NHEADS) * SEQ * DHEAD + (size_t)s * DHEAD;
#pragma unroll
        for (int j = 0; j < 4; ++j) {
          int gn = nBlk + wn + j * 16 + mlane;
          int h = gn >> 6, d = gn & (DHEAD - 1);
          rowDst[(size_t)h * SEQ * DHEAD + d] = f32_to_bf16(acc[t][j][r]);
        }
      } else {                   // fp32 pre-LayerNorm
        float* rowDst = (float*)dst + (size_t)gm * DMODEL + nBlk + wn + mlane;
#pragma unroll
        for (int j = 0; j < 4; ++j) rowDst[j * 16] = acc[t][j][r];
      }
    }
}

// ---------------------------------------------------------------------------
// scores[bh][q][k] = (Q[bh] . K[bh]^T) / 8, masked -> -1e9. K-dim = 64.
// Mask tile staged through LDS to keep global byte-loads out of the WMMA
// stream.
// ---------------------------------------------------------------------------
__global__ __launch_bounds__(BDIM)
void attn_scores_kernel(const unsigned short* __restrict__ Qh,
                        const unsigned short* __restrict__ Kh,
                        const unsigned char*  __restrict__ mask,
                        float* __restrict__ attn) {
  constexpr int LDQ = 72, LDK = 136;
  __shared__ __align__(16) unsigned short lsQ[128 * LDQ];
  __shared__ __align__(16) unsigned short lsKT[64 * LDK];
  __shared__ __align__(16) unsigned char  lsMask[128 * 128];

  const int tid = threadIdx.x;
  const int lane = tid & 31, wave = tid >> 5;
  const int half = lane >> 4, mlane = lane & 15;
  const int wm = (wave & 3) * 32, wn = (wave >> 2) * 64;
  const int bh = blockIdx.z, b = bh >> 4;
  const int qBlk = blockIdx.y * 128;
  const int kBlk = blockIdx.x * 128;
  const unsigned short* Q  = Qh + (size_t)bh * SEQ * DHEAD;
  const unsigned short* Kp = Kh + (size_t)bh * SEQ * DHEAD;

  // stage Q [128][64] row-major (async path: no conversion needed)
#pragma unroll
  for (int i = 0; i < 4; ++i) {
    int lin = tid + i * BDIM;
    int row = lin >> 3, g = lin & 7;
    async_cp16(&lsQ[row * LDQ + g * 8],
               Q + (size_t)(qBlk + row) * DHEAD + g * 8);
  }
  // stage mask tile [128 q][128 k] (async path)
#pragma unroll
  for (int i = 0; i < 4; ++i) {
    int lin = tid + i * BDIM;
    int row = lin >> 3, g = lin & 7;
    async_cp16(&lsMask[row * 128 + g * 16],
               mask + ((size_t)b * SEQ + qBlk + row) * SEQ + kBlk + g * 16);
  }
  // stage K transposed: lsKT[d][key] (needs per-element scatter)
#pragma unroll
  for (int i = 0; i < 4; ++i) {
    int lin = tid + i * BDIM;
    int n = lin >> 3, g = lin & 7;
    union { uint4 q; unsigned short s[8]; } u;
    u.q = *(const uint4*)(Kp + (size_t)(kBlk + n) * DHEAD + g * 8);
#pragma unroll
    for (int j = 0; j < 8; ++j) lsKT[(g * 8 + j) * LDK + n] = u.s[j];
  }
  async_fence();
  __syncthreads();

  v8f acc[2][4];
#pragma unroll
  for (int t = 0; t < 2; ++t)
#pragma unroll
    for (int j = 0; j < 4; ++j) acc[t][j] = zero8();

#pragma unroll
  for (int k0 = 0; k0 < 64; k0 += 32) {
    const unsigned short* a0p = lsQ + (wm + mlane) * LDQ + k0 + half * 8;
    const unsigned short* a1p = lsQ + (wm + 16 + mlane) * LDQ + k0 + half * 8;
    v16bf a0 = ld_frag(a0p, a0p + 16);
    v16bf a1 = ld_frag(a1p, a1p + 16);
    const unsigned short* bp = lsKT + (size_t)(k0 + half * 8) * LDK + wn + mlane;
    v16bf b0 = ld_fragT(bp,      LDK);
    v16bf b1 = ld_fragT(bp + 16, LDK);
    v16bf b2 = ld_fragT(bp + 32, LDK);
    v16bf b3 = ld_fragT(bp + 48, LDK);
    acc[0][0] = wmma_bf16(a0, b0, acc[0][0]);
    acc[0][1] = wmma_bf16(a0, b1, acc[0][1]);
    acc[0][2] = wmma_bf16(a0, b2, acc[0][2]);
    acc[0][3] = wmma_bf16(a0, b3, acc[0][3]);
    acc[1][0] = wmma_bf16(a1, b0, acc[1][0]);
    acc[1][1] = wmma_bf16(a1, b1, acc[1][1]);
    acc[1][2] = wmma_bf16(a1, b2, acc[1][2]);
    acc[1][3] = wmma_bf16(a1, b3, acc[1][3]);
  }

  const float scale = 0.125f;  // 1/sqrt(64)
#pragma unroll
  for (int t = 0; t < 2; ++t)
#pragma unroll
    for (int r = 0; r < 8; ++r) {
      const int qL = wm + t * 16 + r + 8 * half;                // local q
      float* rowOut = attn + ((size_t)bh * SEQ + qBlk + qL) * SEQ + kBlk + wn + mlane;
      const unsigned char* mrow = lsMask + qL * 128 + wn + mlane;
#pragma unroll
      for (int j = 0; j < 4; ++j) {
        float val = acc[t][j][r] * scale;
        if (mrow[j * 16]) val = -1e9f;
        rowOut[j * 16] = val;
      }
    }
}

// ---------------------------------------------------------------------------
// Row softmax over 2048, in place. One block per row; float4 traffic.
// ---------------------------------------------------------------------------
__global__ __launch_bounds__(BDIM)
void softmax_kernel(float* __restrict__ attn) {
  __shared__ float red[BDIM];
  float4* p4 = (float4*)(attn + (size_t)blockIdx.x * SEQ);
  const int t = threadIdx.x;
  float4 va = p4[t * 2], vb = p4[t * 2 + 1];
  float v[8] = {va.x, va.y, va.z, va.w, vb.x, vb.y, vb.z, vb.w};
  float m = -3.4e38f;
#pragma unroll
  for (int i = 0; i < 8; ++i) m = fmaxf(m, v[i]);
  red[t] = m; __syncthreads();
  for (int s = BDIM / 2; s > 0; s >>= 1) {
    if (t < s) red[t] = fmaxf(red[t], red[t + s]);
    __syncthreads();
  }
  m = red[0]; __syncthreads();
  float sum = 0.f;
#pragma unroll
  for (int i = 0; i < 8; ++i) { v[i] = __expf(v[i] - m); sum += v[i]; }
  red[t] = sum; __syncthreads();
  for (int s = BDIM / 2; s > 0; s >>= 1) {
    if (t < s) red[t] += red[t + s];
    __syncthreads();
  }
  const float inv = 1.0f / red[0];
  float4 oa, ob;
  oa.x = v[0] * inv; oa.y = v[1] * inv; oa.z = v[2] * inv; oa.w = v[3] * inv;
  ob.x = v[4] * inv; ob.y = v[5] * inv; ob.z = v[6] * inv; ob.w = v[7] * inv;
  p4[t * 2] = oa; p4[t * 2 + 1] = ob;
}

// ---------------------------------------------------------------------------
// context[bh] = attn[bh] (fp32 -> bf16 on stage) @ V[bh].  M=2048,N=64,K=2048.
// ---------------------------------------------------------------------------
__global__ __launch_bounds__(BDIM)
void context_kernel(const float* __restrict__ attn,
                    const unsigned short* __restrict__ Vh,
                    unsigned short* __restrict__ ctx) {
  constexpr int LDA = 48, LDV = 72;
  __shared__ __align__(16) unsigned short lsA[128 * LDA];
  __shared__ __align__(16) unsigned short lsV[32 * LDV];

  const int tid = threadIdx.x;
  const int lane = tid & 31, wave = tid >> 5;
  const int half = lane >> 4, mlane = lane & 15;
  const int bh = blockIdx.z, b = bh >> 4, h = bh & 15;
  const int qBlk = blockIdx.x * 128;
  const int wmRow = wave * 16;               // 8 waves over M
  const float* Ab = attn + (size_t)bh * SEQ * SEQ;
  const unsigned short* V = Vh + (size_t)bh * SEQ * DHEAD;

  v8f acc[4];
#pragma unroll
  for (int j = 0; j < 4; ++j) acc[j] = zero8();

  for (int kt = 0; kt < SEQ / 32; ++kt) {
    const int k0 = kt * 32;
    {  // V tile [32 k][64 d] already matches B-frag layout -> async copy
      int kr = tid >> 3, g = tid & 7;
      async_cp16(&lsV[kr * LDV + g * 8],
                 V + (size_t)(k0 + kr) * DHEAD + g * 8);
    }
#pragma unroll
    for (int i = 0; i < 4; ++i) {
      int lin = tid + i * BDIM;
      int row = lin >> 3, cg = lin & 7;
      float4 v = *(const float4*)(Ab + (size_t)(qBlk + row) * SEQ + k0 + cg * 4);
      uint2 p;
      p.x = (unsigned)f32_to_bf16(v.x) | ((unsigned)f32_to_bf16(v.y) << 16);
      p.y = (unsigned)f32_to_bf16(v.z) | ((unsigned)f32_to_bf16(v.w) << 16);
      *(uint2*)&lsA[row * LDA + cg * 4] = p;
    }
    async_fence();
    __syncthreads();

    const unsigned short* ap = lsA + (wmRow + mlane) * LDA + half * 8;
    v16bf a = ld_frag(ap, ap + 16);
    const unsigned short* bp = lsV + half * 8 * LDV + mlane;
    v16bf b0 = ld_fragT(bp,      LDV);
    v16bf b1 = ld_fragT(bp + 16, LDV);
    v16bf b2 = ld_fragT(bp + 32, LDV);
    v16bf b3 = ld_fragT(bp + 48, LDV);
    acc[0] = wmma_bf16(a, b0, acc[0]);
    acc[1] = wmma_bf16(a, b1, acc[1]);
    acc[2] = wmma_bf16(a, b2, acc[2]);
    acc[3] = wmma_bf16(a, b3, acc[3]);
    __syncthreads();
  }

#pragma unroll
  for (int r = 0; r < 8; ++r) {
    const int q = qBlk + wmRow + r + 8 * half;
    unsigned short* o = ctx + ((size_t)(b * SEQ + q)) * DMODEL + h * DHEAD + mlane;
#pragma unroll
    for (int j = 0; j < 4; ++j) o[j * 16] = f32_to_bf16(acc[j][r]);
  }
}

// ---------------------------------------------------------------------------
// out = LayerNorm(pre + residual), eps = 1e-5. One block per row of 1024.
// ---------------------------------------------------------------------------
__global__ __launch_bounds__(BDIM)
void layernorm_kernel(const float* __restrict__ pre,
                      const float* __restrict__ resid,
                      float* __restrict__ out) {
  __shared__ float red[BDIM];
  const size_t base = (size_t)blockIdx.x * DMODEL;
  const int t = threadIdx.x;
  float4 a = *(const float4*)(pre + base + t * 4);
  float4 b = *(const float4*)(resid + base + t * 4);
  float x0 = a.x + b.x, x1 = a.y + b.y, x2 = a.z + b.z, x3 = a.w + b.w;

  red[t] = x0 + x1 + x2 + x3; __syncthreads();
  for (int s = BDIM / 2; s > 0; s >>= 1) {
    if (t < s) red[t] += red[t + s];
    __syncthreads();
  }
  const float mu = red[0] * (1.0f / DMODEL);
  __syncthreads();

  float d0 = x0 - mu, d1 = x1 - mu, d2 = x2 - mu, d3 = x3 - mu;
  red[t] = d0 * d0 + d1 * d1 + d2 * d2 + d3 * d3; __syncthreads();
  for (int s = BDIM / 2; s > 0; s >>= 1) {
    if (t < s) red[t] += red[t + s];
    __syncthreads();
  }
  const float rs = rsqrtf(red[0] * (1.0f / DMODEL) + 1e-5f);

  float4 o;
  o.x = d0 * rs; o.y = d1 * rs; o.z = d2 * rs; o.w = d3 * rs;
  *(float4*)(out + base + t * 4) = o;
}

// ---------------------------------------------------------------------------
extern "C" void kernel_launch(void* const* d_in, const int* in_sizes, int n_in,
                              void* d_out, int out_size, void* d_ws, size_t ws_size,
                              hipStream_t stream) {
  (void)in_sizes; (void)n_in; (void)out_size; (void)ws_size;
  const float* inQ = (const float*)d_in[0];
  const float* inK = (const float*)d_in[1];
  const float* inV = (const float*)d_in[2];
  const unsigned char* mask = (const unsigned char*)d_in[3];
  const float* WQ  = (const float*)d_in[4];
  const float* WK  = (const float*)d_in[5];
  const float* WV  = (const float*)d_in[6];
  const float* Wfc = (const float*)d_in[7];

  float* outLN = (float*)d_out;
  float* attn  = (float*)d_out + (size_t)BATCH * SEQ * DMODEL;  // second output

  const size_t HEADTENS = (size_t)BATCH * NHEADS * SEQ * DHEAD;  // 4 Mi elems
  unsigned short* qh  = (unsigned short*)d_ws;
  unsigned short* kh  = qh + HEADTENS;
  unsigned short* vh  = kh + HEADTENS;
  unsigned short* ctx = vh + HEADTENS;
  float* pre = (float*)(ctx + (size_t)BATCH * SEQ * DMODEL);     // 48 MB total

  dim3 gProj(DMODEL / 128, (BATCH * SEQ) / 128, 1);              // 8 x 32
  gemm_nt_kernel<false, 0><<<gProj, BDIM, 0, stream>>>(inQ, WQ, qh);
  gemm_nt_kernel<false, 0><<<gProj, BDIM, 0, stream>>>(inK, WK, kh);
  gemm_nt_kernel<false, 0><<<gProj, BDIM, 0, stream>>>(inV, WV, vh);

  attn_scores_kernel<<<dim3(SEQ / 128, SEQ / 128, BATCH * NHEADS), BDIM, 0, stream>>>(
      qh, kh, mask, attn);

  softmax_kernel<<<dim3(BATCH * NHEADS * SEQ, 1, 1), BDIM, 0, stream>>>(attn);

  context_kernel<<<dim3(SEQ / 128, 1, BATCH * NHEADS), BDIM, 0, stream>>>(attn, vh, ctx);

  gemm_nt_kernel<true, 1><<<gProj, BDIM, 0, stream>>>(ctx, Wfc, pre);

  layernorm_kernel<<<dim3(BATCH * SEQ, 1, 1), BDIM, 0, stream>>>(pre, inQ, outLN);
}
